// LinearDuelingHead_23467701305394
// MI455X (gfx1250) — matrix operand order repaired and checked by reading, not compiled, not run on hardware
//
#include <hip/hip_runtime.h>
#include <hip/hip_bf16.h>

// ---------------------------------------------------------------------------
// CDNA5 (gfx1250) dueling-head fused pipeline:
//   - v_wmma_f32_16x16x32_bf16 for all GEMMs (bf16 A/B, f32 accumulate)
//   - Tensor Data Mover stages packed weights into LDS once per workgroup
//   - run-combined global_atomic_add_f32 segment sums (sorted index)
//   - all A/B fragment traffic via b128 loads, packed bf16 converts
// ---------------------------------------------------------------------------

typedef __attribute__((ext_vector_type(16))) __bf16   bf16x16;
typedef __attribute__((ext_vector_type(2)))  __bf16   bf16x2;
typedef __attribute__((ext_vector_type(8)))  float    floatx8;
typedef __attribute__((ext_vector_type(4)))  float    f32x4;
typedef __attribute__((ext_vector_type(4)))  unsigned v4u;
typedef __attribute__((ext_vector_type(4)))  int      v4i;
typedef __attribute__((ext_vector_type(8)))  int      v8i;

#define DH 128
#define DIN 64
#define SEGS 250000

// LDS layout (dwords) for kernel 1
#define W_VINW 0
#define W_VINP 4096
#define W_AW0  8192
#define W_AP0  12288
#define W_AW1  16384
#define K1_WDW 24576            // 96 KB of packed weights
#define A0_STRIDE 132           // padded 16x128 tile: conflict-free LDS banks
#define A0_DW (16 * A0_STRIDE)  // 2112 dwords per wave

#if defined(__has_builtin)
#if __has_builtin(__builtin_amdgcn_tensor_load_to_lds) && __has_builtin(__builtin_amdgcn_s_wait_tensorcnt)
#define HAVE_TDM 1
#endif
#if __has_builtin(__builtin_amdgcn_cvt_pk_bf16_f32)
#define HAVE_PKCVT 1
#endif
#endif
#ifndef HAVE_TDM
#define HAVE_TDM 0
#endif
#ifndef HAVE_PKCVT
#define HAVE_PKCVT 0
#endif

// Write fragment element pair [2i,2i+1] from two f32 via one packed convert.
__device__ __forceinline__ void setpair(bf16x16& d, int i, float lo, float hi) {
#if HAVE_PKCVT
    bf16x2 p = __builtin_amdgcn_cvt_pk_bf16_f32(lo, hi);
    d[2 * i]     = p[0];
    d[2 * i + 1] = p[1];
#else
    d[2 * i]     = (__bf16)lo;
    d[2 * i + 1] = (__bf16)hi;
#endif
}

__device__ __forceinline__ unsigned packbf(float lo, float hi) {
#if HAVE_PKCVT
    union { bf16x2 v; unsigned u; } t;
    t.v = __builtin_amdgcn_cvt_pk_bf16_f32(lo, hi);
    return t.u;
#else
    union { __bf16 b[2]; unsigned u; } t;
    t.b[0] = (__bf16)lo;
    t.b[1] = (__bf16)hi;
    return t.u;
#endif
}

__device__ __forceinline__ floatx8 zero8() {
    floatx8 c = {0.f, 0.f, 0.f, 0.f, 0.f, 0.f, 0.f, 0.f};
    return c;
}

// Convert 8 consecutive f32 (16B-aligned) into A-fragment elements [8g..8g+7].
// Two b128 loads + 4 packed converts.
__device__ __forceinline__ void abuild8(bf16x16& dst, int g, const float* src8) {
    f32x4 v0 = ((const f32x4*)src8)[0];
    f32x4 v1 = ((const f32x4*)src8)[1];
    setpair(dst, 4 * g + 0, v0[0], v0[1]);
    setpair(dst, 4 * g + 1, v0[2], v0[3]);
    setpair(dst, 4 * g + 2, v1[0], v1[1]);
    setpair(dst, 4 * g + 3, v1[2], v1[3]);
}

// WMMA with A in registers and B fetched from LDS as two b128 loads.
__device__ __forceinline__ floatx8 wmma_ld(bf16x16 a, const unsigned* bsrc, floatx8 c) {
    union { v4u q[2]; bf16x16 v; } B;
    const v4u* p = (const v4u*)bsrc;   // packed layout is 32B-aligned
    B.q[0] = p[0];
    B.q[1] = p[1];
    return __builtin_amdgcn_wmma_f32_16x16x32_bf16(
        false, a, false, B.v, (short)0, c, false, false);
}

#if HAVE_TDM
// TDM: DMA ndw dwords from global to LDS (1-row 2D tile, data_size=4B).
// D# per CDNA5 ISA. ndw must be < 65536.
// 6-arg builtin on this toolchain: (v4u, v8i, v4i, v4i, v8i, i32).
__device__ __forceinline__ void tdm_load_lds(const void* g, unsigned ldsOff, unsigned ndw) {
    unsigned long long ga = (unsigned long long)g;
    v4u g0 = { 1u,                                   // count=1, user descriptor
               ldsOff,                               // lds_addr
               (unsigned)(ga & 0xFFFFFFFFu),         // global_addr[31:0]
               (unsigned)(((ga >> 32) & 0x1FFFFFFu) | (2u << 30)) }; // [56:32] | type=2
    v8i g1 = { (int)(2u << 16),                      // data_size=2 (4 bytes)
               (int)((ndw & 0xFFFFu) << 16),         // tensor_dim0[15:0]
               (int)(((ndw >> 16) & 0xFFFFu) | (1u << 16)), // tensor_dim0[31:16], tensor_dim1=1
               (int)((ndw & 0xFFFFu) << 16),         // tile_dim0 = ndw
               (int)1,                               // tile_dim1 = 1
               (int)ndw,                             // tensor_dim0_stride[31:0]
               0, 0 };
    v4i z4 = {0, 0, 0, 0};
    v8i z8 = {0, 0, 0, 0, 0, 0, 0, 0};
    __builtin_amdgcn_tensor_load_to_lds(g0, g1, z4, z4, z8, 0);
}
#endif

// Stage `ndw` dwords of packed weights into LDS (whole block participates).
__device__ __forceinline__ void stage_weights(unsigned* smem, const unsigned* gsrc, unsigned ndw) {
#if HAVE_TDM
    if ((threadIdx.x >> 5) == 0) {
        tdm_load_lds(gsrc, (unsigned)(unsigned long long)(size_t)smem, ndw);
        __builtin_amdgcn_s_wait_tensorcnt(0);
    }
#else
    for (unsigned t = threadIdx.x; t < ndw; t += blockDim.x) smem[t] = gsrc[t];
#endif
    __syncthreads();
}

// ---------------------------------------------------------------------------
// Pack fp32 weight W[K][128] into bf16 B-matrix VGPR-layout tiles.
// Tile (kt,nt): lane half=lane/16, n=lane%16; dword i holds K=kt*32+8*half+2i+8*(i>=4),+1.
// dst offset = ((kt*8+nt)*32+lane)*8 + i  -> wave reads 1KB contiguous.
// ---------------------------------------------------------------------------
__global__ void kpack_b(const float* __restrict__ W, unsigned* __restrict__ dst, int KT) {
    int t = blockIdx.x * blockDim.x + threadIdx.x;
    int total = KT * 8 * 32 * 8;
    if (t >= total) return;
    int i    = t & 7;
    int lane = (t >> 3) & 31;
    int nt   = (t >> 8) & 7;
    int kt   = t >> 11;
    int half = lane >> 4, nlo = lane & 15;
    int k = kt * 32 + 8 * half + 2 * i + 8 * (i >> 2);
    int n = nt * 16 + nlo;
    dst[t] = packbf(W[k * DH + n], W[(k + 1) * DH + n]);
}

// ---------------------------------------------------------------------------
// Kernel 1: per-row streams, 8 waves/block, 16 rows/wave, weights in LDS.
// ---------------------------------------------------------------------------
__global__ __launch_bounds__(256) void k1_rows(
    const float* __restrict__ x, const long long* __restrict__ idx, int N,
    const unsigned* __restrict__ wblk,     // 96KB packed: vinW|vinP|aW0|aP0|aW1
    const float* __restrict__ v_in_b, const float* __restrict__ a_b0,
    const float* __restrict__ a_b1,
    const float* __restrict__ a_lin_W, const float* __restrict__ a_lin_b,
    float* __restrict__ agg, float* __restrict__ seg_sum,
    float* __restrict__ cnt, float* __restrict__ adv_out)
{
    extern __shared__ unsigned smem[];
    const int wave = threadIdx.x >> 5;
    const int lane = threadIdx.x & 31;

    stage_weights(smem, wblk, K1_WDW);

    const int tile = blockIdx.x * 8 + wave;
    const int rowBase = tile * 16;
    if (rowBase >= N) return;

    const int half = lane >> 4, nl = lane & 15;
    const int m = nl;
    float* a0w = (float*)(smem + K1_WDW) + wave * A0_DW;

    int segs[8];
#pragma unroll
    for (int r = 0; r < 8; r++) {
        int row = rowBase + r + 8 * half;
        segs[r] = (row < N) ? (int)idx[row] : -1;
    }

    // x tile in bf16 A-matrix layout (two K-tiles), b128 loads
    bf16x16 xa[2];
    {
        int row = rowBase + m; if (row >= N) row = N - 1;
        const float* xr = x + (long long)row * DIN;
#pragma unroll
        for (int kt = 0; kt < 2; kt++)
#pragma unroll
        for (int g = 0; g < 2; g++)
            abuild8(xa[kt], g, xr + kt * 32 + 16 * g + 8 * half);
    }

    // ---------------- value in-block + run-combined segment atomics -------
    for (int nt = 0; nt < 8; nt++) {
        floatx8 c1 = zero8(), c2 = zero8();
#pragma unroll
        for (int kt = 0; kt < 2; kt++) {
            c1 = wmma_ld(xa[kt], smem + W_VINW + ((kt * 8 + nt) * 32 + lane) * 8, c1);
            c2 = wmma_ld(xa[kt], smem + W_VINP + ((kt * 8 + nt) * 32 + lane) * 8, c2);
        }
        const float bv = v_in_b[nt * 16 + nl];
        const int col = nt * 16 + nl;
        float acc = 0.f;
#pragma unroll
        for (int r = 0; r < 8; r++) {
            float hv = fmaxf(c1[r] + bv, 0.f) + c2[r];
            acc += hv;
            bool flush = (r == 7) || (segs[r + 1] != segs[r]);
            if (flush) {
                if (segs[r] >= 0)
                    atomicAdd(&agg[(long long)segs[r] * DH + col], acc);
                acc = 0.f;
            }
        }
    }

    // ---------------- advantage block 0 -> LDS (D->A transpose) -----------
    for (int nt = 0; nt < 8; nt++) {
        floatx8 c1 = zero8(), c2 = zero8();
#pragma unroll
        for (int kt = 0; kt < 2; kt++) {
            c1 = wmma_ld(xa[kt], smem + W_AW0 + ((kt * 8 + nt) * 32 + lane) * 8, c1);
            c2 = wmma_ld(xa[kt], smem + W_AP0 + ((kt * 8 + nt) * 32 + lane) * 8, c2);
        }
        const float b0 = a_b0[nt * 16 + nl];
#pragma unroll
        for (int r = 0; r < 8; r++) {
            float v = fmaxf(c1[r] + b0, 0.f) + c2[r];
            a0w[(r + 8 * half) * A0_STRIDE + nt * 16 + nl] = v;
        }
    }

    // Build A fragments for a0 ONCE (hoisted; b128 LDS loads)
    bf16x16 aA[4];
#pragma unroll
    for (int kt = 0; kt < 4; kt++)
#pragma unroll
    for (int g = 0; g < 2; g++)
        abuild8(aA[kt], g, a0w + m * A0_STRIDE + kt * 32 + 16 * g + 8 * half);

    // ---------------- advantage block 1 + head partial dot ----------------
    float padv[8];
#pragma unroll
    for (int r = 0; r < 8; r++) padv[r] = 0.f;

    for (int nt = 0; nt < 8; nt++) {
        floatx8 c = zero8();
#pragma unroll
        for (int kt = 0; kt < 4; kt++)
            c = wmma_ld(aA[kt], smem + W_AW1 + ((kt * 8 + nt) * 32 + lane) * 8, c);
        const float b1  = a_b1[nt * 16 + nl];
        const float alw = a_lin_W[nt * 16 + nl];
#pragma unroll
        for (int r = 0; r < 8; r++) {
            float res = a0w[(r + 8 * half) * A0_STRIDE + nt * 16 + nl];
            float v = fmaxf(c[r] + b1, 0.f) + res;
            padv[r] += v * alw;
        }
    }

#pragma unroll
    for (int mask = 1; mask < 16; mask <<= 1)
#pragma unroll
        for (int r = 0; r < 8; r++)
            padv[r] += __shfl_xor(padv[r], mask, 32);

    if (nl == 0) {
        const float ab = a_lin_b[0];
        float acc = 0.f, accc = 0.f;
#pragma unroll
        for (int r = 0; r < 8; r++) {
            int row = rowBase + r + 8 * half;
            if (row < N) {
                float v = padv[r] + ab;
                adv_out[row] = v;
                acc += v; accc += 1.f;
            }
            bool flush = (r == 7) || (segs[r + 1] != segs[r]);
            if (flush) {
                if (segs[r] >= 0) {
                    atomicAdd(&seg_sum[segs[r]], acc);
                    atomicAdd(&cnt[segs[r]], accc);
                }
                acc = 0.f; accc = 0.f;
            }
        }
    }
}

// ---------------------------------------------------------------------------
// Kernel 2: segment stream, 8 waves/block, 16 segments/wave, v_out_W in LDS.
// ---------------------------------------------------------------------------
__global__ __launch_bounds__(256) void k2_segs(
    const float* __restrict__ agg,
    const unsigned* __restrict__ p_voutW, const float* __restrict__ v_out_b,
    const float* __restrict__ v_lin_W, const float* __restrict__ v_lin_b,
    float* __restrict__ sv, int S)
{
    extern __shared__ unsigned smem[];
    const int wave = threadIdx.x >> 5;
    const int lane = threadIdx.x & 31;

    stage_weights(smem, p_voutW, 4 * 8 * 32 * 8);   // 32 KB

    const int tile = blockIdx.x * 8 + wave;
    const int segBase = tile * 16;
    if (segBase >= S) return;

    const int half = lane >> 4, nl = lane & 15;
    const int m = nl;

    bf16x16 aA[4];
    {
        int srow = segBase + m; if (srow >= S) srow = S - 1;
        const float* ar = agg + (long long)srow * DH;
#pragma unroll
        for (int kt = 0; kt < 4; kt++)
#pragma unroll
        for (int g = 0; g < 2; g++)
            abuild8(aA[kt], g, ar + kt * 32 + 16 * g + 8 * half);
    }

    float psv[8];
#pragma unroll
    for (int r = 0; r < 8; r++) psv[r] = 0.f;

    for (int nt = 0; nt < 8; nt++) {
        floatx8 c = zero8();
#pragma unroll
        for (int kt = 0; kt < 4; kt++)
            c = wmma_ld(aA[kt], smem + ((kt * 8 + nt) * 32 + lane) * 8, c);
        const float b  = v_out_b[nt * 16 + nl];
        const float vw = v_lin_W[nt * 16 + nl];
#pragma unroll
        for (int r = 0; r < 8; r++) {
            int srow = segBase + r + 8 * half;
            float res = (srow < S) ? agg[(long long)srow * DH + nt * 16 + nl] : 0.f;
            float h2 = fmaxf(c[r] + b, 0.f) + res;
            psv[r] += h2 * vw;
        }
    }

#pragma unroll
    for (int mask = 1; mask < 16; mask <<= 1)
#pragma unroll
        for (int r = 0; r < 8; r++)
            psv[r] += __shfl_xor(psv[r], mask, 32);

    if (nl == 0) {
        const float vb = v_lin_b[0];
#pragma unroll
        for (int r = 0; r < 8; r++) {
            int srow = segBase + r + 8 * half;
            if (srow < S) sv[srow] = psv[r] + vb;
        }
    }
}

// ---------------------------------------------------------------------------
// Kernel 3: gather-back and combine.
// ---------------------------------------------------------------------------
__global__ void k3_final(const long long* __restrict__ idx,
                         const float* __restrict__ adv, const float* __restrict__ sv,
                         const float* __restrict__ seg_sum, const float* __restrict__ cnt,
                         float* __restrict__ out, int N, int copyIdx)
{
    int i = blockIdx.x * blockDim.x + threadIdx.x;
    if (i >= N) return;
    long long iv = idx[i];
    int s = (int)iv;
    float c = cnt[s]; c = (c < 1.f) ? 1.f : c;
    out[i] = sv[s] + adv[i] - seg_sum[s] / c;
    if (copyIdx) ((long long*)(out + N))[i] = iv;
}

// ---------------------------------------------------------------------------
extern "C" void kernel_launch(void* const* d_in, const int* in_sizes, int n_in,
                              void* d_out, int out_size, void* d_ws, size_t ws_size,
                              hipStream_t stream) {
    const float*     x       = (const float*)d_in[0];
    const long long* index   = (const long long*)d_in[1];
    const float*     v_in_W  = (const float*)d_in[2];
    const float*     v_in_b  = (const float*)d_in[3];
    const float*     v_in_P  = (const float*)d_in[4];
    const float*     v_out_W = (const float*)d_in[5];
    const float*     v_out_b = (const float*)d_in[6];
    const float*     v_lin_W = (const float*)d_in[7];
    const float*     v_lin_b = (const float*)d_in[8];
    const float*     a_W0    = (const float*)d_in[9];
    const float*     a_b0    = (const float*)d_in[10];
    const float*     a_P0    = (const float*)d_in[11];
    const float*     a_W1    = (const float*)d_in[12];
    const float*     a_b1    = (const float*)d_in[13];
    const float*     a_lin_W = (const float*)d_in[14];
    const float*     a_lin_b = (const float*)d_in[15];

    const int N = in_sizes[0] / DIN;
    const int S = SEGS;
    float* out = (float*)d_out;

    char* ws = (char*)d_ws;
    size_t o = 0;
    float* agg     = (float*)(ws + o); o += (size_t)S * DH * 4;
    float* seg_sum = (float*)(ws + o); o += (size_t)S * 4;
    float* cnt     = (float*)(ws + o); o += (size_t)S * 4;
    float* adv     = (float*)(ws + o); o += (size_t)N * 4;
    float* sv      = (float*)(ws + o); o += (size_t)S * 4;
    const int SZ64  = 2 * 8 * 32 * 8;
    const int SZ128 = 4 * 8 * 32 * 8;
    // k1 weight block must be contiguous (one TDM copy): vinW|vinP|aW0|aP0|aW1
    unsigned* p_vinW  = (unsigned*)(ws + o); o += (size_t)SZ64 * 4;
    unsigned* p_vinP  = (unsigned*)(ws + o); o += (size_t)SZ64 * 4;
    unsigned* p_aW0   = (unsigned*)(ws + o); o += (size_t)SZ64 * 4;
    unsigned* p_aP0   = (unsigned*)(ws + o); o += (size_t)SZ64 * 4;
    unsigned* p_aW1   = (unsigned*)(ws + o); o += (size_t)SZ128 * 4;
    unsigned* p_voutW = (unsigned*)(ws + o); o += (size_t)SZ128 * 4;
    (void)ws_size; (void)n_in;

    (void)hipMemsetAsync(agg, 0, (size_t)S * DH * 4 + (size_t)S * 8, stream);

    kpack_b<<<(SZ64  + 127) / 128, 128, 0, stream>>>(v_in_W,  p_vinW,  2);
    kpack_b<<<(SZ64  + 127) / 128, 128, 0, stream>>>(v_in_P,  p_vinP,  2);
    kpack_b<<<(SZ64  + 127) / 128, 128, 0, stream>>>(a_W0,    p_aW0,   2);
    kpack_b<<<(SZ64  + 127) / 128, 128, 0, stream>>>(a_P0,    p_aP0,   2);
    kpack_b<<<(SZ128 + 127) / 128, 128, 0, stream>>>(a_W1,    p_aW1,   4);
    kpack_b<<<(SZ128 + 127) / 128, 128, 0, stream>>>(v_out_W, p_voutW, 4);

    int rowTiles = (N + 15) / 16;
    size_t k1_lds = (size_t)K1_WDW * 4 + 8 * (size_t)A0_DW * 4;   // 96KB + 66KB
    k1_rows<<<(rowTiles + 7) / 8, 256, k1_lds, stream>>>(
        x, index, N, p_vinW, v_in_b, a_b0, a_b1, a_lin_W, a_lin_b,
        agg, seg_sum, cnt, adv);

    int segTiles = (S + 15) / 16;
    size_t k2_lds = (size_t)SZ128 * 4;                            // 32KB
    k2_segs<<<(segTiles + 7) / 8, 256, k2_lds, stream>>>(
        agg, p_voutW, v_out_b, v_lin_W, v_lin_b, sv, S);

    int copyIdx = (out_size >= 3 * N) ? 1 : 0;
    k3_final<<<(N + 255) / 256, 256, 0, stream>>>(index, adv, sv, seg_sum, cnt,
                                                  out, N, copyIdx);
}
